// SimpleRNN_4071628996712
// MI455X (gfx1250) — matrix operand order repaired and checked by reading, not compiled, not run on hardware
//
#include <hip/hip_runtime.h>

typedef __attribute__((ext_vector_type(16))) _Float16 v16h;
typedef __attribute__((ext_vector_type(8)))  _Float16 v8h;
typedef __attribute__((ext_vector_type(8)))  float    v8f;

#define RNN_B 4096
#define RNN_T 1024
#define RNN_H 16

// ---- f16 tanh builtin: probe both plausible spellings ----
#if __has_builtin(__builtin_amdgcn_tanh_f16)
#define TANH16(x) __builtin_amdgcn_tanh_f16(x)
#define HAVE_TANH16 1
#elif __has_builtin(__builtin_amdgcn_tanhh)
#define TANH16(x) __builtin_amdgcn_tanhh(x)
#define HAVE_TANH16 1
#else
#define HAVE_TANH16 0
#endif

#if HAVE_TANH16 && __has_builtin(__builtin_amdgcn_wmma_f16_16x16x32_f16)
#define F16_PATH 1     // chain: wmma(f16 C/D) -> v_tanh_f16 -> wmma
#else
#define F16_PATH 0     // chain: wmma(f32 C/D) -> v_tanh_f32 -> cvt_pk -> wmma
#endif

#if !F16_PATH
#if __has_builtin(__builtin_amdgcn_tanhf)
#define PRE_SCALE 1.0f
__device__ __forceinline__ float act(float x) {
  return __builtin_amdgcn_tanhf(x);
}
#else
#define PRE_SCALE 2.8853900817779268f  // 2*log2(e), folded into WMMA inputs
__device__ __forceinline__ float act(float x) {
  float e = __builtin_amdgcn_exp2f(x);
  return 1.0f - 2.0f * __builtin_amdgcn_rcpf(e + 1.0f);
}
#endif
#else
#define PRE_SCALE 1.0f
#endif

// K-remap (no cross-lane traffic on the recurrence chain):
//   B[k,n]=h[n,k] k=0..7 (lane n, elems0..7, K=0..7);
//   B[16+j,n]=h[n,8+j]   (lane n+16, elems0..7, K=16..23); K=8..15,24..31 zero.
//   A[m,k]=W_hh[m,k] k=0..7; A[m,16+j]=W_hh[m,8+j]; rest zero
//   => lane half h holds W_hh[row, 8h..8h+7] in elems 0..7; elems 8..15 zero.
// C/D element e (f32 vgpr r=e, or f16 half e) <-> (hid e+8*khalf, batch n),
// identical to the B-frag element map, so the rebuild is lane-local.

__global__ __launch_bounds__(32) void rnn_wmma_kernel(
    const float* __restrict__ x,     // [B,T,1]
    const float* __restrict__ W_ih,  // [H,1]
    const float* __restrict__ b_ih,  // [H]
    const float* __restrict__ W_hh,  // [H,H] row-major
    const float* __restrict__ b_hh,  // [H]
    const float* __restrict__ fc_W,  // [1,H]
    const float* __restrict__ fc_b,  // [1]
    float* __restrict__ out)         // [B,T,1]
{
  const int lane  = threadIdx.x;
  const int col   = lane & 15;
  const int khalf = lane >> 4;
  const int b     = blockIdx.x * 16 + col;

  // ---- A fragment: remapped W_hh. Lane half h, elems 0..7: W_hh[row, 8h+j]; elems 8..15: zero ----
  v16h aF = (v16h)(_Float16)0;
  {
#pragma unroll
    for (int j = 0; j < 8; ++j)
      aF[j] = (_Float16)(PRE_SCALE * W_hh[col * RNN_H + 8 * khalf + j]);
  }

  float fcw[8];
#pragma unroll
  for (int r = 0; r < 8; ++r) fcw[r] = fc_W[r + 8 * khalf];
  const float fcb = fc_b[0];
  const bool  low = (lane < 16);

  // ---- B fragment: hidden state, h0 = 0. Elems 8..15 stay zero forever. ----
  v16h bF = (v16h)(_Float16)0;

  const float* xp = x   + (size_t)b * RNN_T;
  float*       op = out + (size_t)b * RNN_T;

#if F16_PATH
  // packed-f16 per-lane constants for the C (xp) operand
  v8h wihh, biash;
#pragma unroll
  for (int r = 0; r < 8; ++r) {
    int h    = r + 8 * khalf;
    wihh[r]  = (_Float16)W_ih[h];
    biash[r] = (_Float16)(b_ih[h] + b_hh[h]);
  }

  for (int t0 = 0; t0 < RNN_T; t0 += 4) {
    const float4 xv = *(const float4*)(xp + t0);  // 4 timesteps of x (b128)
    float o[4];
#pragma unroll
    for (int u = 0; u < 4; ++u) {
      const _Float16 xth = (_Float16)((&xv.x)[u]);

      // C = xp_t in packed f16 (off-chain, v_pk_fma_f16)
      v8h c;
#pragma unroll
      for (int r = 0; r < 8; ++r) c[r] = (_Float16)(xth * wihh[r] + biash[r]);

      // D(f16) = A * Hs + xp_t
      v8h d = __builtin_amdgcn_wmma_f16_16x16x32_f16(
          /*neg_a=*/false, aF, /*neg_b=*/false, bF,
          /*c_mod=*/(short)0, c, /*reuse_a=*/false, /*reuse_b=*/false);

      // h = tanh(D) written straight into the B fragment (lane-local, f16)
#pragma unroll
      for (int r = 0; r < 8; ++r) bF[r] = TANH16(d[r]);

      // FC output (off the recurrence chain), mixed-precision dot
      float p = 0.f;
#pragma unroll
      for (int r = 0; r < 8; ++r) p = __builtin_fmaf((float)bF[r], fcw[r], p);
      const float po = __shfl_xor(p, 16, 32);
      o[u] = p + po + fcb;
    }
    if (low) {
      *(float4*)(op + t0) = make_float4(o[0], o[1], o[2], o[3]);
    }
  }
#else
  float wih[8], bias[8];
#pragma unroll
  for (int r = 0; r < 8; ++r) {
    int h   = r + 8 * khalf;
    wih[r]  = PRE_SCALE * W_ih[h];
    bias[r] = PRE_SCALE * (b_ih[h] + b_hh[h]);
  }

  for (int t0 = 0; t0 < RNN_T; t0 += 4) {
    const float4 xv = *(const float4*)(xp + t0);
    float o[4];
#pragma unroll
    for (int u = 0; u < 4; ++u) {
      const float xt = (&xv.x)[u];
      v8f c;
#pragma unroll
      for (int r = 0; r < 8; ++r) c[r] = __builtin_fmaf(xt, wih[r], bias[r]);
      v8f d = __builtin_amdgcn_wmma_f32_16x16x32_f16(
          false, aF, false, bF, (short)0, c, false, false);
      float hv[8];
#pragma unroll
      for (int r = 0; r < 8; ++r) hv[r] = act(d[r]);
#pragma unroll
      for (int r = 0; r < 8; ++r) bF[r] = (_Float16)hv[r];
      float p = 0.f;
#pragma unroll
      for (int r = 0; r < 8; ++r) p = __builtin_fmaf(hv[r], fcw[r], p);
      const float po = __shfl_xor(p, 16, 32);
      o[u] = p + po + fcb;
    }
    if (low) {
      *(float4*)(op + t0) = make_float4(o[0], o[1], o[2], o[3]);
    }
  }
#endif
}

extern "C" void kernel_launch(void* const* d_in, const int* in_sizes, int n_in,
                              void* d_out, int out_size, void* d_ws, size_t ws_size,
                              hipStream_t stream) {
  const float* x    = (const float*)d_in[0];
  const float* W_ih = (const float*)d_in[1];
  const float* b_ih = (const float*)d_in[2];
  const float* W_hh = (const float*)d_in[3];
  const float* b_hh = (const float*)d_in[4];
  const float* fc_W = (const float*)d_in[5];
  const float* fc_b = (const float*)d_in[6];
  float* out = (float*)d_out;

  dim3 grid(RNN_B / 16);  // 256 waves, one 16-batch tile each
  dim3 block(32);         // one wave32 per workgroup
  hipLaunchKernelGGL(rnn_wmma_kernel, grid, block, 0, stream,
                     x, W_ih, b_ih, W_hh, b_hh, fc_W, fc_b, out);
}
